// KnowledgeRNN_19808389169798
// MI455X (gfx1250) — compile-verified, matrix operand order checked by reading, not run on hardware
//
#include <hip/hip_runtime.h>
#include <math.h>

// Problem dims
#define S_LEN 2048
#define NTOK  32000
#define EDIM  512
#define HDIM  1024
#define QSZ   40
#define VSZ   80
#define NKBK  16384
#define QIN   1536          // H + E
#define G4    4096          // 4H
#define FEAT  1616          // E + VS + H
#define FPAD  1632          // FEAT padded to multiple of 32 (and of 96) for WMMA K-loop

#define NWG   32            // persistent workgroups for the recurrent scan
#define BLK   256
#define GLANES (NWG*BLK)    // 8192

typedef __attribute__((ext_vector_type(16))) _Float16 v16h;
typedef __attribute__((ext_vector_type(8)))  _Float16 v8h;
typedef __attribute__((ext_vector_type(8)))  float    v8f;
typedef unsigned int u32x4 __attribute__((ext_vector_type(4)));
typedef int          i32x8 __attribute__((ext_vector_type(8)));
typedef int          i32x4 __attribute__((ext_vector_type(4)));

// ---------------------------------------------------------------- utilities
__device__ __forceinline__ float sigm(float x) { return 1.f / (1.f + __expf(-x)); }

// Sense-reversing grid barrier (all NWG workgroups must be resident).
__device__ __forceinline__ void grid_barrier(unsigned* cnt, unsigned* gen) {
  __threadfence();
  __syncthreads();
  if (threadIdx.x == 0) {
    unsigned g = __hip_atomic_load(gen, __ATOMIC_RELAXED, __HIP_MEMORY_SCOPE_AGENT);
    unsigned a = __hip_atomic_fetch_add(cnt, 1u, __ATOMIC_ACQ_REL, __HIP_MEMORY_SCOPE_AGENT);
    if (a == NWG - 1u) {
      __hip_atomic_store(cnt, 0u, __ATOMIC_RELAXED, __HIP_MEMORY_SCOPE_AGENT);
      __hip_atomic_fetch_add(gen, 1u, __ATOMIC_ACQ_REL, __HIP_MEMORY_SCOPE_AGENT);
    } else {
      while (__hip_atomic_load(gen, __ATOMIC_ACQUIRE, __HIP_MEMORY_SCOPE_AGENT) == g)
        __builtin_amdgcn_s_sleep(2);
    }
  }
  __syncthreads();
}

// --------------------------------------------------------------- TDM helpers
// Issue a 2D Tensor-Data-Mover load of a [rows x cols] f16 tile (row stride
// ld_elems) from global into LDS at byte offset lds_off. D# per CDNA5 ISA §8.
// This toolchain exposes the 6-arg builtin form:
//   (uint32x4 g0, int32x8 g1, int32x4 g2, int32x4 g3, int32x8 gx, int cpol)
#if __has_builtin(__builtin_amdgcn_tensor_load_to_lds)
#define HAVE_TDM 1
__device__ __forceinline__ void tdm_load_2d(unsigned lds_off, const _Float16* gptr,
                                            int rows, int cols, int ld_elems,
                                            int tensor_rows) {
  unsigned long long ga = (unsigned long long)(size_t)gptr;
  unsigned td0 = (unsigned)ld_elems;            // tensor_dim0 (row length, elems)
  unsigned td1 = (unsigned)tensor_rows;         // tensor_dim1 (total rows)
  unsigned long long st0 = (unsigned long long)(unsigned)ld_elems; // dim0 stride
  u32x4 g0;
  g0.x = 1u;                                              // count=1, user D#
  g0.y = lds_off;                                         // lds_addr (bytes)
  g0.z = (unsigned)(ga & 0xffffffffu);                    // global_addr[31:0]
  g0.w = (unsigned)((ga >> 32) & 0x01ffffffu) | (2u << 30); // addr[56:32] | type=2
  i32x8 g1;
  g1[0] = 0x00010000;                                     // data_size=1 (2B); no mcast
  g1[1] = (int)((td0 & 0xffffu) << 16);                   // tensor_dim0[15:0]
  g1[2] = (int)((td0 >> 16) | ((td1 & 0xffffu) << 16));   // td0[31:16] | td1[15:0]
  g1[3] = (int)((td1 >> 16) | (((unsigned)cols & 0xffffu) << 16)); // td1 hi | tile_dim0
  g1[4] = (int)((unsigned)rows & 0xffffu);                // tile_dim1 (tile_dim2=0)
  g1[5] = (int)(st0 & 0xffffffffu);                       // tensor_dim0_stride lo
  g1[6] = (int)((st0 >> 32) & 0xffffu);                   // stride hi | dim1_stride=0
  g1[7] = 0;
  i32x4 gz  = {0, 0, 0, 0};
  i32x8 gz8 = {0, 0, 0, 0, 0, 0, 0, 0};
  __builtin_amdgcn_tensor_load_to_lds(g0, g1, gz, gz, gz8, 0);
}
__device__ __forceinline__ void tdm_wait() { __builtin_amdgcn_s_wait_tensorcnt(0); }
#else
#define HAVE_TDM 0
#endif

// ---------------------------------------------------------------- prep kernels
// Gather embeddings -> feats f16 (cols [0,E)), zero the K-pad cols [FEAT,FPAD)
__global__ void k_embed(const int* __restrict__ input, const float* __restrict__ encw,
                        _Float16* __restrict__ feats) {
  int idx = blockIdx.x * BLK + threadIdx.x;
  if (idx >= S_LEN * FPAD) return;
  int s = idx / FPAD, c = idx % FPAD;
  if (c < EDIM)       feats[idx] = (_Float16)encw[(size_t)input[s] * EDIM + c];
  else if (c >= FEAT) feats[idx] = (_Float16)0.f;
}

// w_dec f32 [NT,1616] -> f16 [NT,1632] (zero-padded K)
__global__ void k_conv_wdec(const float* __restrict__ wdec, _Float16* __restrict__ w16) {
  long idx = (long)blockIdx.x * BLK + threadIdx.x;
  if (idx >= (long)NTOK * FPAD) return;
  int n = (int)(idx / FPAD), c = (int)(idx % FPAD);
  w16[idx] = (c < FEAT) ? (_Float16)wdec[(size_t)n * FEAT + c] : (_Float16)0.f;
}

// wq1 rows [H, H+E) transposed -> f16 [QIN(out n)][EDIM(k)]
__global__ void k_wq1x(const float* __restrict__ wq1, _Float16* __restrict__ o) {
  int idx = blockIdx.x * BLK + threadIdx.x;
  if (idx >= QIN * EDIM) return;
  int n = idx / EDIM, k = idx % EDIM;
  o[idx] = (_Float16)wq1[(size_t)(HDIM + k) * QIN + n];
}

// w_ih first E cols -> f16 [G4][EDIM]
__global__ void k_wihx(const float* __restrict__ wih, _Float16* __restrict__ o) {
  int idx = blockIdx.x * BLK + threadIdx.x;
  if (idx >= G4 * EDIM) return;
  int n = idx / EDIM, k = idx % EDIM;
  o[idx] = (_Float16)wih[(size_t)n * (EDIM + VSZ) + k];
}

// w_hh [G4][H] -> transposed f32 [H][G4]
__global__ void k_whhT(const float* __restrict__ whh, float* __restrict__ o) {
  int idx = blockIdx.x * BLK + threadIdx.x;
  if (idx >= HDIM * G4) return;
  int i = idx / G4, g = idx % G4;
  o[idx] = whh[(size_t)g * HDIM + i];
}

// w_ih value cols -> transposed f32 [VS][G4]
__global__ void k_wihv(const float* __restrict__ wih, float* __restrict__ o) {
  int idx = blockIdx.x * BLK + threadIdx.x;
  if (idx >= VSZ * G4) return;
  int v = idx / G4, g = idx % G4;
  o[idx] = wih[(size_t)g * (EDIM + VSZ) + EDIM + v];
}

// b_ih + b_hh ; zero hx/cx/barrier state (re-init every call: no cross-call state)
__global__ void k_misc(const float* __restrict__ bih, const float* __restrict__ bhh,
                       float* __restrict__ bsum, float* __restrict__ hx,
                       float* __restrict__ cx, unsigned* __restrict__ bars) {
  int idx = blockIdx.x * BLK + threadIdx.x;
  if (idx < G4) bsum[idx] = bih[idx] + bhh[idx];
  if (idx < HDIM) { hx[idx] = 0.f; cx[idx] = 0.f; }
  if (idx < 16) bars[idx] = 0u;
}

// ------------------------------------------------- WMMA fragment helpers
__device__ __forceinline__ v16h frag16(const _Float16* p) {
  v8h lo = *(const v8h*)p;
  v8h hi = *(const v8h*)(p + 16);
  v16h r;
#pragma unroll
  for (int i = 0; i < 8; ++i) { r[i] = lo[i]; r[i + 8] = hi[i]; }
  return r;
}

// ---------------------------------------------------------------- WMMA GEMM
// C[m,n] = sum_k A[m,k]*B[n,k] + bias[n].   A,B f16 row-major (K contiguous).
// grid.x = N/64, grid.y = M/32 ; block = 256 (8 waves, 2(M)x4(N) of 16x16 tiles).
__global__ __launch_bounds__(256) void wmma_gemm(
    const _Float16* __restrict__ A, const _Float16* __restrict__ B,
    const float* __restrict__ bias, float* __restrict__ C,
    int K, int lda, int ldb, int ldc)
{
  int wave = threadIdx.x >> 5;
  int lane = threadIdx.x & 31;
  int m0 = blockIdx.y * 32 + (wave & 1) * 16;
  int n0 = blockIdx.x * 64 + (wave >> 1) * 16;
  int krun = (lane >> 4) * 8;       // ISA 16-bit A/B fragment: two 8-elem K runs

  const _Float16* aBase = A + (size_t)(m0 + (lane & 15)) * lda + krun;
  const _Float16* bBase = B + (size_t)(n0 + (lane & 15)) * ldb + krun;

  v8f acc = {};
  for (int k = 0; k < K; k += 32) {
    __builtin_prefetch(aBase + k + 64, 0, 1);   // -> global_prefetch_b8
    __builtin_prefetch(bBase + k + 64, 0, 1);
    v16h a = frag16(aBase + k);
    v16h b = frag16(bBase + k);
    acc = __builtin_amdgcn_wmma_f32_16x16x32_f16(false, a, false, b,
                                                 (short)0, acc, false, false);
  }
  int n = n0 + (lane & 15);
  int r0 = m0 + ((lane >> 4) ? 8 : 0);
  float bv = bias ? bias[n] : 0.f;
#pragma unroll
  for (int r = 0; r < 8; ++r)
    C[(size_t)(r0 + r) * ldc + n] = acc[r] + bv;
}

// ------------------------------------------- WMMA GEMM with TDM -> LDS staging
// Double-buffered: Tensor Data Mover DMAs a 32xkstep A-slab and 64xkstep B-slab
// into LDS; waves consume fragments from LDS (ds_load_b128) while the next slab
// is in flight (TENSORcnt).  kstep must divide K; kstep <= 96, multiple of 32.
#define KSTEP_MAX 96
__global__ __launch_bounds__(256) void wmma_gemm_tdm(
    const _Float16* __restrict__ A, const _Float16* __restrict__ B,
    const float* __restrict__ bias, float* __restrict__ C,
    int K, int kstep, int lda, int ldb, int ldc, int Mtot, int Ntot)
{
  __shared__ __align__(16) _Float16 ldsA[2][32 * KSTEP_MAX];
  __shared__ __align__(16) _Float16 ldsB[2][64 * KSTEP_MAX];

  int wave = threadIdx.x >> 5;
  int lane = threadIdx.x & 31;
  int wm = wave & 1, wn = wave >> 1;
  int mBlk = blockIdx.y * 32, nBlk = blockIdx.x * 64;
  int nstages = K / kstep;

  int ar = wm * 16 + (lane & 15);       // local A row
  int br = wn * 16 + (lane & 15);       // local B row
  int krun = (lane >> 4) * 8;

#if HAVE_TDM
  if (threadIdx.x == 0) {
    tdm_load_2d((unsigned)(size_t)&ldsA[0][0], A + (size_t)mBlk * lda, 32, kstep, lda, Mtot);
    tdm_load_2d((unsigned)(size_t)&ldsB[0][0], B + (size_t)nBlk * ldb, 64, kstep, ldb, Ntot);
  }
#endif
  v8f acc = {};
  for (int s = 0; s < nstages; ++s) {
    int buf = s & 1;
#if HAVE_TDM
    if (threadIdx.x == 0) tdm_wait();      // slab s landed in LDS
    __syncthreads();                        // publish to all waves
    if (threadIdx.x == 0 && s + 1 < nstages) {
      int k = (s + 1) * kstep;             // prefetch slab s+1 into other buffer
      tdm_load_2d((unsigned)(size_t)&ldsA[buf ^ 1][0], A + (size_t)mBlk * lda + k,
                  32, kstep, lda, Mtot);
      tdm_load_2d((unsigned)(size_t)&ldsB[buf ^ 1][0], B + (size_t)nBlk * ldb + k,
                  64, kstep, ldb, Ntot);
    }
#else
    { // cooperative fallback copy (toolchains without the TDM builtin)
      int k = s * kstep;
      for (int e = threadIdx.x; e < 32 * kstep / 8; e += BLK) {
        int rr = e / (kstep / 8), cc = (e % (kstep / 8)) * 8;
        *(v8h*)&ldsA[buf][rr * kstep + cc] = *(const v8h*)(A + (size_t)(mBlk + rr) * lda + k + cc);
      }
      for (int e = threadIdx.x; e < 64 * kstep / 8; e += BLK) {
        int rr = e / (kstep / 8), cc = (e % (kstep / 8)) * 8;
        *(v8h*)&ldsB[buf][rr * kstep + cc] = *(const v8h*)(B + (size_t)(nBlk + rr) * ldb + k + cc);
      }
      __syncthreads();
    }
#endif
    for (int kk = 0; kk < kstep; kk += 32) {
      v16h a = frag16(&ldsA[buf][ar * kstep + kk + krun]);
      v16h b = frag16(&ldsB[buf][br * kstep + kk + krun]);
      acc = __builtin_amdgcn_wmma_f32_16x16x32_f16(false, a, false, b,
                                                   (short)0, acc, false, false);
    }
    __syncthreads();                        // all waves done with buf before reuse
  }
  int n = nBlk + wn * 16 + (lane & 15);
  int r0 = mBlk + wm * 16 + ((lane >> 4) ? 8 : 0);
  float bv = bias ? bias[n] : 0.f;
#pragma unroll
  for (int r = 0; r < 8; ++r)
    C[(size_t)(r0 + r) * ldc + n] = acc[r] + bv;
}

// ---------------------------------------------------------------- recurrent scan
// Persistent NWG x BLK kernel; 6 grid barriers per step.
__global__ __launch_bounds__(BLK) void rnn_persistent(
    const float* __restrict__ wq1,   // [QIN][QIN] f32, rows [0,H) used
    const float* __restrict__ wq2,   // [QIN][QS]
    const float* __restrict__ bq2,   // [QS]
    const float* __restrict__ kbk,   // [NKB][QS]
    const float* __restrict__ kbv,   // [NKB][VS]
    const float* __restrict__ qpre,  // [S][QIN]  (x-part of wq1 + bq1, from GEMM)
    const float* __restrict__ xih,   // [S][G4]   (x-part of w_ih + b_ih + b_hh)
    const float* __restrict__ whhT,  // [H][G4]
    const float* __restrict__ wihv,  // [VS][G4]
    float* __restrict__ hx, float* __restrict__ cx,
    float* __restrict__ qh, float* __restrict__ scr, float* __restrict__ gates,
    float* __restrict__ wgmax, float* __restrict__ wgsum, float* __restrict__ valp,
    _Float16* __restrict__ feats, unsigned* __restrict__ bars)
{
  unsigned* cnt = bars;
  unsigned* gen = bars + 1;
  const int gl  = blockIdx.x * BLK + threadIdx.x;
  const int tid = threadIdx.x;

  __shared__ float red[BLK];
  __shared__ float lq[6 * QSZ];      // also reused as 3x80 value partials
  __shared__ float lQ[QSZ];
  __shared__ float lval[VSZ];

  for (int t = 0; t < S_LEN; ++t) {
    // -------- Phase A: q_h = tanh(qpre[t] + hx @ wq1[0:H,:])  (4 lanes/output)
    if (gl < QIN * 4) {
      int j = gl >> 2, seg = gl & 3;
      float p = (seg == 0) ? qpre[(size_t)t * QIN + j] : 0.f;
      int i0 = seg * 256;
      for (int i = i0; i < i0 + 256; ++i) p += hx[i] * wq1[(size_t)i * QIN + j];
      p += __shfl_xor(p, 1, 32);
      p += __shfl_xor(p, 2, 32);
      if (seg == 0) qh[j] = tanhf(p);
    }
    grid_barrier(cnt, gen);

    // -------- Phase B: q (redundant per WG) then scores + per-WG max
    if (tid < 6 * QSZ) {
      int j = tid % QSZ, seg = tid / QSZ;
      float p = 0.f;
      int i0 = seg * 256;
      for (int i = i0; i < i0 + 256; ++i) p += qh[i] * wq2[(size_t)i * QSZ + j];
      lq[tid] = p;
    }
    __syncthreads();
    if (tid < QSZ) {
      float p = bq2[tid];
#pragma unroll
      for (int s2 = 0; s2 < 6; ++s2) p += lq[s2 * QSZ + tid];
      lQ[tid] = p;
    }
    __syncthreads();

    float lmax = -3.4e38f;
#pragma unroll
    for (int r = 0; r < 2; ++r) {
      int k = gl + r * GLANES;
      const float* kr = kbk + (size_t)k * QSZ;
      float s = 0.f;
#pragma unroll 8
      for (int j = 0; j < QSZ; ++j) s += kr[j] * lQ[j];
      scr[k] = s;
      lmax = fmaxf(lmax, s);
    }
    red[tid] = lmax; __syncthreads();
    for (int s2 = 128; s2 > 0; s2 >>= 1) {
      if (tid < s2) red[tid] = fmaxf(red[tid], red[tid + s2]);
      __syncthreads();
    }
    if (tid == 0) wgmax[blockIdx.x] = red[0];
    grid_barrier(cnt, gen);

    // -------- Phase B2: exp(score - gmax) + per-WG sum
    float gmax = -3.4e38f;
#pragma unroll
    for (int w = 0; w < NWG; ++w) gmax = fmaxf(gmax, wgmax[w]);
    float lsum = 0.f;
#pragma unroll
    for (int r = 0; r < 2; ++r) {
      int k = gl + r * GLANES;
      float e = __expf(scr[k] - gmax);
      scr[k] = e;
      lsum += e;
    }
    red[tid] = lsum; __syncthreads();
    for (int s2 = 128; s2 > 0; s2 >>= 1) {
      if (tid < s2) red[tid] += red[tid + s2];
      __syncthreads();
    }
    if (tid == 0) wgsum[blockIdx.x] = red[0];
    grid_barrier(cnt, gen);

    // -------- Phase B3: per-WG partial of val = attn @ kb_values (512 keys/WG)
    if (tid < 3 * VSZ) {
      int v = tid % VSZ, seg = tid / VSZ;
      float p = 0.f;
      int kbeg = blockIdx.x * (NKBK / NWG);
      for (int k = kbeg + seg; k < kbeg + NKBK / NWG; k += 3)
        p += scr[k] * kbv[(size_t)k * VSZ + v];
      lq[tid] = p;
    }
    __syncthreads();
    if (tid < VSZ)
      valp[blockIdx.x * VSZ + tid] = lq[tid] + lq[VSZ + tid] + lq[2 * VSZ + tid];
    grid_barrier(cnt, gen);

    // -------- Phase C: reduce val (redundant, fixed order), gates matvec
    float gsum = 0.f;
#pragma unroll
    for (int w = 0; w < NWG; ++w) gsum += wgsum[w];
    float inv = 1.f / gsum;
    if (tid < VSZ) {
      float v = 0.f;
#pragma unroll
      for (int w = 0; w < NWG; ++w) v += valp[w * VSZ + tid];
      v *= inv;
      lval[tid] = v;
      if (blockIdx.x == 0) feats[(size_t)t * FPAD + EDIM + tid] = (_Float16)v;
    }
    __syncthreads();
    {
      int g = gl >> 1, half = gl & 1;
      float acc = half ? 0.f : xih[(size_t)t * G4 + g];
      int i0 = half * 512;
      for (int i = i0; i < i0 + 512; ++i) acc += hx[i] * whhT[(size_t)i * G4 + g];
      int v0 = half * 40;
#pragma unroll 8
      for (int v = v0; v < v0 + 40; ++v) acc += lval[v] * wihv[(size_t)v * G4 + g];
      acc += __shfl_xor(acc, 1, 32);
      if (!half) gates[g] = acc;
    }
    grid_barrier(cnt, gen);

    // -------- Phase D: LSTM pointwise; record PREVIOUS hx into feats
    if (gl < HDIM) {
      int h = gl;
      float ig = gates[h], fg = gates[HDIM + h];
      float gg = gates[2 * HDIM + h], og = gates[3 * HDIM + h];
      float oldh = hx[h];
      feats[(size_t)t * FPAD + EDIM + VSZ + h] = (_Float16)oldh;
      float c  = sigm(fg) * cx[h] + sigm(ig) * tanhf(gg);
      float hn = sigm(og) * tanhf(c);
      cx[h] = c; hx[h] = hn;
    }
    grid_barrier(cnt, gen);
  }
}

// ---------------------------------------------------------------- log_softmax rows
__global__ __launch_bounds__(BLK) void k_logsoftmax(float* __restrict__ out, int n) {
  float* p = out + (size_t)blockIdx.x * n;
  __shared__ float red[BLK];
  int tid = threadIdx.x;
  float mx = -3.4e38f;
  for (int i = tid; i < n; i += BLK) mx = fmaxf(mx, p[i]);
  red[tid] = mx; __syncthreads();
  for (int s = 128; s > 0; s >>= 1) { if (tid < s) red[tid] = fmaxf(red[tid], red[tid + s]); __syncthreads(); }
  mx = red[0]; __syncthreads();
  float sum = 0.f;
  for (int i = tid; i < n; i += BLK) sum += __expf(p[i] - mx);
  red[tid] = sum; __syncthreads();
  for (int s = 128; s > 0; s >>= 1) { if (tid < s) red[tid] += red[tid + s]; __syncthreads(); }
  float lse = mx + logf(red[0]);
  for (int i = tid; i < n; i += BLK) p[i] = p[i] - lse;
}

// ---------------------------------------------------------------- launch
extern "C" void kernel_launch(void* const* d_in, const int* in_sizes, int n_in,
                              void* d_out, int out_size, void* d_ws, size_t ws_size,
                              hipStream_t stream) {
  (void)in_sizes; (void)n_in; (void)out_size; (void)ws_size;
  const int*   input = (const int*)  d_in[0];
  const float* encw  = (const float*)d_in[1];
  const float* wq1   = (const float*)d_in[2];
  const float* bq1   = (const float*)d_in[3];
  const float* wq2   = (const float*)d_in[4];
  const float* bq2   = (const float*)d_in[5];
  const float* kbk   = (const float*)d_in[6];
  const float* kbv   = (const float*)d_in[7];
  const float* w_ih  = (const float*)d_in[8];
  const float* b_ih  = (const float*)d_in[9];
  const float* w_hh  = (const float*)d_in[10];
  const float* b_hh  = (const float*)d_in[11];
  const float* w_dec = (const float*)d_in[12];
  const float* b_dec = (const float*)d_in[13];
  float* out = (float*)d_out;

  char* ws = (char*)d_ws;
  size_t off = 0;
  auto alloc = [&](size_t bytes) -> char* {
    off = (off + 255) & ~(size_t)255;
    char* p = ws + off; off += bytes; return p;
  };
  _Float16* feats  = (_Float16*)alloc((size_t)S_LEN * FPAD * 2);
  _Float16* wdec16 = (_Float16*)alloc((size_t)NTOK  * FPAD * 2);
  _Float16* wq1x16 = (_Float16*)alloc((size_t)QIN   * EDIM * 2);
  _Float16* wihx16 = (_Float16*)alloc((size_t)G4    * EDIM * 2);
  float* qpre  = (float*)alloc((size_t)S_LEN * QIN * 4);
  float* xih   = (float*)alloc((size_t)S_LEN * G4  * 4);
  float* whhT  = (float*)alloc((size_t)HDIM  * G4  * 4);
  float* wihv  = (float*)alloc((size_t)VSZ   * G4  * 4);
  float* bsum  = (float*)alloc(G4 * 4);
  float* hx    = (float*)alloc(HDIM * 4);
  float* cx    = (float*)alloc(HDIM * 4);
  float* qh    = (float*)alloc(QIN * 4);
  float* scr   = (float*)alloc(NKBK * 4);
  float* gates = (float*)alloc(G4 * 4);
  float* wgmax = (float*)alloc(NWG * 4);
  float* wgsum = (float*)alloc(NWG * 4);
  float* valp  = (float*)alloc(NWG * VSZ * 4);
  unsigned* bars = (unsigned*)alloc(64);

  // prep
  k_embed    <<<(S_LEN * FPAD + BLK - 1) / BLK, BLK, 0, stream>>>(input, encw, feats);
  k_conv_wdec<<<(int)(((long)NTOK * FPAD + BLK - 1) / BLK), BLK, 0, stream>>>(w_dec, wdec16);
  k_wq1x     <<<(QIN * EDIM + BLK - 1) / BLK, BLK, 0, stream>>>(wq1, wq1x16);
  k_wihx     <<<(G4 * EDIM + BLK - 1) / BLK, BLK, 0, stream>>>(w_ih, wihx16);
  k_whhT     <<<(HDIM * G4 + BLK - 1) / BLK, BLK, 0, stream>>>(w_hh, whhT);
  k_wihv     <<<(VSZ * G4 + BLK - 1) / BLK, BLK, 0, stream>>>(w_ih, wihv);
  k_misc     <<<(G4 + BLK - 1) / BLK, BLK, 0, stream>>>(b_ih, b_hh, bsum, hx, cx, bars);

  // x-dependent projections as WMMA GEMMs (bias folded in), direct-from-L2
  wmma_gemm<<<dim3(QIN / 64, S_LEN / 32), 256, 0, stream>>>(
      feats, wq1x16, bq1, qpre, EDIM, FPAD, EDIM, QIN);
  wmma_gemm<<<dim3(G4 / 64, S_LEN / 32), 256, 0, stream>>>(
      feats, wihx16, bsum, xih, EDIM, FPAD, EDIM, G4);

  // sequential knowledge-LSTM scan (persistent grid)
  rnn_persistent<<<NWG, BLK, 0, stream>>>(
      wq1, wq2, bq2, kbk, kbv, qpre, xih, whhT, wihv,
      hx, cx, qh, scr, gates, wgmax, wgsum, valp, feats, bars);

  // decoder GEMM: logits = feats @ w_dec^T + b_dec, TDM-staged LDS double buffer
  // K = 1632 = 17 stages of 96
  wmma_gemm_tdm<<<dim3(NTOK / 64, S_LEN / 32), 256, 0, stream>>>(
      feats, wdec16, b_dec, out, FPAD, 96, FPAD, FPAD, NTOK, S_LEN, NTOK);

  // row-wise log_softmax in place
  k_logsoftmax<<<S_LEN, BLK, 0, stream>>>(out, NTOK);
}
